// DWNBlock_5214090297874
// MI455X (gfx1250) — matrix operand order repaired and checked by reading, not compiled, not run on hardware
//
#include <hip/hip_runtime.h>
#include <cstdint>
#include <cstddef>

// ---------------------------------------------------------------- constants
static constexpr int BATCH  = 16;
static constexpr int LSEQ   = 8192;
static constexpr int DM     = 256;   // d_model
static constexpr int DS     = 256;   // d_state
static constexpr int DFF    = 1024;
static constexpr int MTOT   = BATCH * LSEQ;        // 131072 rows
static constexpr int NCHUNK = 64;                  // scan chunks per sequence
static constexpr int TCH    = LSEQ / NCHUNK;       // 128 steps per chunk
static constexpr int KCAT   = 3 * DS;              // 768: [st_re | st_im | x]

typedef __attribute__((ext_vector_type(16))) __bf16 v16bf;
typedef __attribute__((ext_vector_type(8)))  float  v8f;

// ---------------------------------------------------------------- prep: lambda
__global__ void prep_lam(const float* __restrict__ nu_log,
                         const float* __restrict__ theta_log,
                         float* __restrict__ lam_re, float* __restrict__ lam_im,
                         float* __restrict__ lamT_re, float* __restrict__ lamT_im) {
    int s = threadIdx.x;                         // 256 threads
    float la    = expf(-expf(nu_log[s]));        // |lambda|
    float theta = expf(theta_log[s]);
    float lr = la * cosf(theta);
    float li = la * sinf(theta);
    lam_re[s] = lr; lam_im[s] = li;
    // lambda^TCH  (TCH = 128 = 2^7): repeated squaring
    float ar = lr, ai = li;
    #pragma unroll
    for (int i = 0; i < 7; ++i) {
        float nr = ar * ar - ai * ai;
        float ni = 2.f * ar * ai;
        ar = nr; ai = ni;
    }
    lamT_re[s] = ar; lamT_im[s] = ai;
}

// ---------------------------------------------------------------- prep: weights -> bf16
static constexpr int R0 = 512 * DM;          // Wbu   (512 x 256)
static constexpr int R1 = R0 + DM * KCAT;    // Wy    (256 x 768) = [C_re | -C_im | D]
static constexpr int R2 = R1 + DFF * DM;     // Wfct  (1024 x 256)
static constexpr int R3 = R2 + DM * DFF;     // Wpjt  (256 x 1024)

__global__ void prep_weights(const float* __restrict__ gamma_log,
                             const float* __restrict__ B_re, const float* __restrict__ B_im,
                             const float* __restrict__ C_re, const float* __restrict__ C_im,
                             const float* __restrict__ Dmat,
                             const float* __restrict__ W_fc, const float* __restrict__ W_proj,
                             __bf16* __restrict__ Wbu, __bf16* __restrict__ Wy,
                             __bf16* __restrict__ Wfct, __bf16* __restrict__ Wpjt) {
    int idx = blockIdx.x * 256 + threadIdx.x;
    if (idx >= R3) return;
    if (idx < R0) {
        int row = idx / DM, d = idx % DM;
        int s = row & (DS - 1);
        float g = expf(gamma_log[s]);
        float v = (row < DS) ? B_re[s * DM + d] : B_im[s * DM + d];
        Wbu[idx] = (__bf16)(g * v);
    } else if (idx < R1) {
        int i = idx - R0;
        int o = i / KCAT, k = i % KCAT;
        float v;
        if (k < DS)            v =  C_re[o * DS + k];
        else if (k < 2 * DS)   v = -C_im[o * DS + (k - DS)];
        else                   v =  Dmat[o * DM + (k - 2 * DS)];
        Wy[i] = (__bf16)v;
    } else if (idx < R2) {
        int i = idx - R1;
        int n = i / DM, k = i % DM;          // W_fc (DM, DFF) -> (DFF, DM)
        Wfct[i] = (__bf16)W_fc[k * DFF + n];
    } else {
        int i = idx - R2;
        int n = i / DFF, k = i % DFF;        // W_proj (DFF, DM) -> (DM, DFF)
        Wpjt[i] = (__bf16)W_proj[k * DM + n];
    }
}

// ---------------------------------------------------------------- x -> bf16 into Acat[:, 512:768]
__global__ void convert_x(const float* __restrict__ x, __bf16* __restrict__ Acat) {
    size_t idx = (size_t)blockIdx.x * 256 + threadIdx.x;
    size_t m = idx >> 8;
    int    d = (int)(idx & 255);
    Acat[m * KCAT + 2 * DS + d] = (__bf16)x[idx];
}

// ---------------------------------------------------------------- scan pass A: per-chunk local scan
__global__ void scan_chunk(const float* __restrict__ Bu,
                           const float* __restrict__ lam_re, const float* __restrict__ lam_im,
                           float* __restrict__ pend_re, float* __restrict__ pend_im) {
    int g  = blockIdx.x * 256 + threadIdx.x;   // (b*NCHUNK + c) * 256 + s
    int s  = g & 255;
    int bc = g >> 8;
    int b  = bc >> 6;
    int c  = bc & (NCHUNK - 1);
    const float lr = lam_re[s], li = lam_im[s];
    const float* p = Bu + ((size_t)(b * LSEQ + c * TCH)) * 512 + s;
    float sr = 0.f, si = 0.f;
    for (int t = 0; t < TCH; ++t) {
        float ur = p[0], ui = p[256];
        float nr = lr * sr - li * si + ur;
        float ni = lr * si + li * sr + ui;
        sr = nr; si = ni;
        p += 512;
    }
    pend_re[g] = sr; pend_im[g] = si;
}

// ---------------------------------------------------------------- scan pass B: carry scan across chunks
__global__ void scan_carry(const float* __restrict__ lamT_re, const float* __restrict__ lamT_im,
                           float* __restrict__ pend_re, float* __restrict__ pend_im) {
    int tid = blockIdx.x * 256 + threadIdx.x;  // B*DS threads
    int s = tid & 255;
    int b = tid >> 8;
    const float lr = lamT_re[s], li = lamT_im[s];
    float cr = 0.f, ci = 0.f;
    for (int c = 0; c < NCHUNK; ++c) {
        int idx = ((b * NCHUNK + c) << 8) + s;
        float pr = pend_re[idx], pi = pend_im[idx];
        float nr = pr + lr * cr - li * ci;
        float ni = pi + lr * ci + li * cr;
        pend_re[idx] = nr; pend_im[idx] = ni;
        cr = nr; ci = ni;
    }
}

// ---------------------------------------------------------------- scan pass C: seeded re-scan, emit bf16
__global__ void scan_emit(const float* __restrict__ Bu,
                          const float* __restrict__ lam_re, const float* __restrict__ lam_im,
                          const float* __restrict__ pend_re, const float* __restrict__ pend_im,
                          __bf16* __restrict__ Acat) {
    int g  = blockIdx.x * 256 + threadIdx.x;
    int s  = g & 255;
    int bc = g >> 8;
    int b  = bc >> 6;
    int c  = bc & (NCHUNK - 1);
    const float lr = lam_re[s], li = lam_im[s];
    float sr = 0.f, si = 0.f;
    if (c > 0) {
        int pidx = ((b * NCHUNK + c - 1) << 8) + s;
        sr = pend_re[pidx]; si = pend_im[pidx];
    }
    const float* p = Bu + ((size_t)(b * LSEQ + c * TCH)) * 512 + s;
    __bf16* q = Acat + ((size_t)(b * LSEQ + c * TCH)) * KCAT + s;
    for (int t = 0; t < TCH; ++t) {
        float ur = p[0], ui = p[256];
        float nr = lr * sr - li * si + ur;
        float ni = lr * si + li * sr + ui;
        sr = nr; si = ni;
        q[0]   = (__bf16)sr;
        q[256] = (__bf16)si;
        p += 512;
        q += KCAT;
    }
}

// ---------------------------------------------------------------- bf16 WMMA GEMM with async-LDS pipeline
// out = A(MxK) * Bw(NxK)^T.  Block tile 128x128, BK=64, double-buffered 64KB LDS.
// LDS tile layout: 128 rows x 64 bf16 (128B/row, 8 x 16B segments), segment index
// XOR-swizzled by (row>>1)&7 so wave-wide b128 reads hit the 2-cycle bank minimum.
// MODE 0: ->f32   MODE 1: ->bf16   MODE 2: +bias,GELU->bf16   MODE 3: +bias,+resid->f32
static constexpr int LDS_TILE  = 128 * 64 * 2;         // 16 KB per tile
static constexpr int LDS_TOTAL = 4 * LDS_TILE;         // 64 KB (A x2, B x2)

template <int MODE>
__global__ __launch_bounds__(256) void gemm_bf16(
        const __bf16* __restrict__ A, int lda,
        const __bf16* __restrict__ Bw,            // (N x K) row-major, ldb = K
        const float* __restrict__ bias,
        const float* __restrict__ resid,
        float* __restrict__ outF, __bf16* __restrict__ outB,
        int N, int K) {
    extern __shared__ char smem[];                // [A0|A1|B0|B1], 16KB each
    const int tid   = threadIdx.x;
    const int wave  = tid >> 5;
    const int lane  = tid & 31;
    const int h     = lane >> 4;
    const int l4    = lane & 15;
    const int waveM = wave >> 2;     // 0..1
    const int waveN = wave & 3;      // 0..3
    const int blockRow = blockIdx.x * 128;
    const int blockCol = blockIdx.y * 128;
    const int rowBase  = blockRow + waveM * 64;
    const int colBase  = blockCol + waveN * 32;

    // issue async global->LDS copies for one K-tile (A and B), 16B per lane per issue
    auto issue_tile = [&](int buf, int k0) {
        #pragma unroll
        for (int i = 0; i < 4; ++i) {
            int idx = tid + i * 256;                 // 0..1023 segment slots
            int row = idx >> 3;
            int seg = idx & 7;
            int sw  = (row >> 1) & 7;
            unsigned loffA = (unsigned)(buf * LDS_TILE + row * 128 + ((seg ^ sw) << 4));
            unsigned long long gaA =
                (unsigned long long)(A + (size_t)(blockRow + row) * lda + k0 + seg * 8);
            asm volatile("global_load_async_to_lds_b128 %0, %1, off"
                         :: "v"(loffA), "v"(gaA) : "memory");
            unsigned loffB = (unsigned)(2 * LDS_TILE + buf * LDS_TILE + row * 128 + ((seg ^ sw) << 4));
            unsigned long long gaB =
                (unsigned long long)(Bw + (size_t)(blockCol + row) * K + k0 + seg * 8);
            asm volatile("global_load_async_to_lds_b128 %0, %1, off"
                         :: "v"(loffB), "v"(gaB) : "memory");
        }
    };

    v8f acc[4][2];
    #pragma unroll
    for (int i = 0; i < 4; ++i)
        #pragma unroll
        for (int j = 0; j < 2; ++j)
            acc[i][j] = (v8f){0.f, 0.f, 0.f, 0.f, 0.f, 0.f, 0.f, 0.f};

    issue_tile(0, 0);
    asm volatile("s_wait_asynccnt 0x0" ::: "memory");
    __syncthreads();

    int buf = 0;
    for (int k0 = 0; k0 < K; k0 += 64) {
        const bool more = (k0 + 64) < K;
        if (more) issue_tile(buf ^ 1, k0 + 64);      // prefetch next tile while computing

        const char* Abase = smem + buf * LDS_TILE;
        const char* Bbase = smem + 2 * LDS_TILE + buf * LDS_TILE;
        #pragma unroll
        for (int ks = 0; ks < 2; ++ks) {             // two K=32 sub-steps per BK=64
            v16bf a[4];
            #pragma unroll
            for (int i = 0; i < 4; ++i) {
                // A 16x32 frag: lane half h holds K[8h..8h+7] and K[16+8h..+7]
                int rt = waveM * 64 + i * 16 + l4;
                int sw = (rt >> 1) & 7;
                const char* rb = Abase + rt * 128;
                int s0 = ks * 4 + h;
                union { v16bf v; uint4 u[2]; } fa;
                fa.u[0] = *(const uint4*)(rb + ((s0 ^ sw) << 4));
                fa.u[1] = *(const uint4*)(rb + (((s0 + 2) ^ sw) << 4));
                a[i] = fa.v;
            }
            v16bf bfr[2];
            #pragma unroll
            for (int j = 0; j < 2; ++j) {
                // B 32x16 frag: lane half h holds K[8h..8h+7] and K[16+8h..+7] of column l4
                int nt = waveN * 32 + j * 16 + l4;
                int sw = (nt >> 1) & 7;
                const char* rb = Bbase + nt * 128;
                int s0 = ks * 4 + h;
                union { v16bf v; uint4 u[2]; } fb;
                fb.u[0] = *(const uint4*)(rb + ((s0 ^ sw) << 4));
                fb.u[1] = *(const uint4*)(rb + (((s0 + 2) ^ sw) << 4));
                bfr[j] = fb.v;
            }
            #pragma unroll
            for (int i = 0; i < 4; ++i)
                #pragma unroll
                for (int j = 0; j < 2; ++j)
                    acc[i][j] = __builtin_amdgcn_wmma_f32_16x16x32_bf16(
                        false, a[i], false, bfr[j], (short)0, acc[i][j], false, false);
        }

        if (more) asm volatile("s_wait_asynccnt 0x0" ::: "memory");
        __syncthreads();
        buf ^= 1;
    }

    // Epilogue. C/D layout: VGPR r -> (m = r + 8h, n = l4) within each 16x16 tile.
    #pragma unroll
    for (int i = 0; i < 4; ++i) {
        #pragma unroll
        for (int j = 0; j < 2; ++j) {
            const int col = colBase + j * 16 + l4;
            float bv = 0.f;
            if (MODE == 2 || MODE == 3) bv = bias[col];
            #pragma unroll
            for (int r = 0; r < 8; ++r) {
                const int row = rowBase + i * 16 + h * 8 + r;
                const size_t o = (size_t)row * N + col;
                float v = acc[i][j][r];
                if (MODE == 0) {
                    outF[o] = v;
                } else if (MODE == 1) {
                    outB[o] = (__bf16)v;
                } else if (MODE == 2) {
                    v += bv;
                    float g = 0.5f * v * (1.f + erff(v * 0.70710678118654752f));
                    outB[o] = (__bf16)g;
                } else { // MODE 3
                    v += bv + resid[o];
                    outF[o] = v;
                }
            }
        }
    }
}

// ---------------------------------------------------------------- launch
extern "C" void kernel_launch(void* const* d_in, const int* in_sizes, int n_in,
                              void* d_out, int out_size, void* d_ws, size_t ws_size,
                              hipStream_t stream) {
    (void)in_sizes; (void)n_in; (void)out_size; (void)ws_size;

    const float* x         = (const float*)d_in[0];
    const float* nu_log    = (const float*)d_in[1];
    const float* theta_log = (const float*)d_in[2];
    const float* gamma_log = (const float*)d_in[3];
    const float* B_re      = (const float*)d_in[4];
    const float* B_im      = (const float*)d_in[5];
    const float* C_re      = (const float*)d_in[6];
    const float* C_im      = (const float*)d_in[7];
    const float* Dmat      = (const float*)d_in[8];
    const float* W_fc      = (const float*)d_in[9];
    const float* b_fc      = (const float*)d_in[10];
    const float* W_proj    = (const float*)d_in[11];
    const float* b_proj    = (const float*)d_in[12];
    float*       out       = (float*)d_out;

    // workspace carve-up (256B aligned)
    char* ws = (char*)d_ws;
    size_t off = 0;
    auto carve = [&](size_t bytes) -> char* {
        off = (off + 255) & ~(size_t)255;
        char* p = ws + off;
        off += bytes;
        return p;
    };
    float*  lam_re  = (float*)carve(DS * sizeof(float));
    float*  lam_im  = (float*)carve(DS * sizeof(float));
    float*  lamT_re = (float*)carve(DS * sizeof(float));
    float*  lamT_im = (float*)carve(DS * sizeof(float));
    __bf16* Wbu     = (__bf16*)carve((size_t)R0 * 2);
    __bf16* Wy      = (__bf16*)carve((size_t)(R1 - R0) * 2);
    __bf16* Wfct    = (__bf16*)carve((size_t)(R2 - R1) * 2);
    __bf16* Wpjt    = (__bf16*)carve((size_t)(R3 - R2) * 2);
    __bf16* Acat    = (__bf16*)carve((size_t)MTOT * KCAT * 2);       // 201 MB
    float*  Bu      = (float*)carve((size_t)MTOT * 512 * 4);         // 268 MB
    __bf16* Zbuf    = (__bf16*)carve((size_t)MTOT * DM * 2);         //  67 MB
    __bf16* Hbuf    = (__bf16*)carve((size_t)MTOT * DFF * 2);        // 268 MB
    float*  pend_re = (float*)carve((size_t)BATCH * DS * NCHUNK * 4);
    float*  pend_im = (float*)carve((size_t)BATCH * DS * NCHUNK * 4);

    // 1) lambda and lambda^TCH
    prep_lam<<<1, 256, 0, stream>>>(nu_log, theta_log, lam_re, lam_im, lamT_re, lamT_im);

    // 2) weight conversion / packing to bf16
    prep_weights<<<(R3 + 255) / 256, 256, 0, stream>>>(
        gamma_log, B_re, B_im, C_re, C_im, Dmat, W_fc, W_proj, Wbu, Wy, Wfct, Wpjt);

    // 3) x -> bf16 into Acat[:, 512:768]
    convert_x<<<(MTOT * DM) / 256, 256, 0, stream>>>(x, Acat);

    // 4) Bu = x_bf16 @ [gB_re | gB_im]^T   (M x 512, f32)
    gemm_bf16<0><<<dim3(MTOT / 128, 512 / 128), 256, LDS_TOTAL, stream>>>(
        Acat + 2 * DS, KCAT, Wbu, nullptr, nullptr, Bu, nullptr, 512, DM);

    // 5) chunked complex scan -> bf16 states into Acat[:, 0:512]
    scan_chunk<<<BATCH * NCHUNK, 256, 0, stream>>>(Bu, lam_re, lam_im, pend_re, pend_im);
    scan_carry<<<(BATCH * DS) / 256, 256, 0, stream>>>(lamT_re, lamT_im, pend_re, pend_im);
    scan_emit<<<BATCH * NCHUNK, 256, 0, stream>>>(Bu, lam_re, lam_im, pend_re, pend_im, Acat);

    // 6) z = [st_re | st_im | x] @ [C_re | -C_im | D]^T  -> bf16
    gemm_bf16<1><<<dim3(MTOT / 128, DM / 128), 256, LDS_TOTAL, stream>>>(
        Acat, KCAT, Wy, nullptr, nullptr, nullptr, Zbuf, DM, KCAT);

    // 7) h = gelu(z @ W_fc + b_fc) -> bf16
    gemm_bf16<2><<<dim3(MTOT / 128, DFF / 128), 256, LDS_TOTAL, stream>>>(
        Zbuf, DM, Wfct, b_fc, nullptr, nullptr, Hbuf, DFF, DM);

    // 8) out = h @ W_proj + b_proj + x -> f32
    gemm_bf16<3><<<dim3(MTOT / 128, DM / 128), 256, LDS_TOTAL, stream>>>(
        Hbuf, DFF, Wpjt, b_proj, x, out, nullptr, DM, DFF);
}
